// ObservationModel_61125974556755
// MI455X (gfx1250) — matrix operand order repaired and checked by reading, not compiled, not run on hardware
//
#include <hip/hip_runtime.h>
#include <hip/hip_bf16.h>

typedef __attribute__((ext_vector_type(16))) __bf16 v16bf;
typedef __attribute__((ext_vector_type(8)))  __bf16 v8bf;
typedef __attribute__((ext_vector_type(8)))  float  v8f;

#define BATCHN 32
#define SEQ    2048
#define DIN    384
#define DH     64
#define DOUTN  64
#define NROW   (BATCHN*SEQ)   // 65536
#define SCOLS  192            // [Bu_re(64) | Bu_im(64) | Du(64)]
#define CHUNK  128
#define NCHUNK (SEQ/CHUNK)    // 16
#define XPAD   392            // 384 + 8 bf16 pad -> 784B row stride (16B aligned, bank-spread)
#define XSPAD  136            // 128 + 8 bf16 pad -> 272B row stride

// workspace byte offsets (all >=32B aligned)
#define OFF_S    0ull                 // 65536*192 bf16 = 25,165,824 B
#define OFF_W    25165824ull          // 12*12*32*16 bf16 = 147,456 B
#define OFF_C    25313280ull          // 4*4*32*16 bf16 = 16,384 B
#define OFF_LAM  25329664ull          // 64 * float4 = 1,024 B
#define OFF_CAR  25330688ull          // 32*16*64 * float2 = 262,144 B
#define OFF_PMAX 25592832ull          // 32*16*64 * float = 131,072 B

// Native bf16 converts: clang emits v_cvt_(pk_)bf16_f32 / v_cvt_f32_bf16
__device__ __forceinline__ float bf2f(__bf16 x) { return (float)x; }
__device__ __forceinline__ __bf16 f2bf(float f) { return (__bf16)f; }

// A-fragment gather from LDS row (row-major bf16): elements 0-7 <- K hsel*8..,
// elements 8-15 <- K 16+hsel*8.. (CDNA5 16-bit A layout, wave32)
__device__ __forceinline__ v16bf ldfragA(const __bf16* base, int hsel) {
  v8bf lo = *(const v8bf*)(base + hsel * 8);
  v8bf hi = *(const v8bf*)(base + 16 + hsel * 8);
  return __builtin_shufflevector(lo, hi, 0,1,2,3,4,5,6,7,8,9,10,11,12,13,14,15);
}

// ---------------------------------------------------------------------------
// k_prep: pack W (384x192) and C (128x64, with -C_im) into WMMA-B lane layout;
// precompute lam and lam^128 per channel.
// Packed entry: [kt][nt][lane][16 bf16]; element e = W(k = kt*32+(lane>=16)*8+(e&7)+(e>=8)*16,
// n = nt*16+(lane&15)) -> each lane's B fragment is one 32-byte load.
// ---------------------------------------------------------------------------
__global__ void k_prep(const float* __restrict__ nu_log, const float* __restrict__ theta_log,
                       const float* __restrict__ gamma_log,
                       const float* __restrict__ B_re, const float* __restrict__ B_im,
                       const float* __restrict__ C_re, const float* __restrict__ C_im,
                       const float* __restrict__ Dm,
                       __bf16* __restrict__ Wp, __bf16* __restrict__ Cp,
                       float4* __restrict__ lamT) {
  const int tid = blockIdx.x * blockDim.x + threadIdx.x;
  const int nth = gridDim.x * blockDim.x;
  if (tid < 64) {
    float nu  = expf(nu_log[tid]);
    float th  = expf(theta_log[tid]);
    float mag = expf(-nu);
    float m128 = expf(-128.f * nu);
    float a128 = 128.f * th;
    lamT[tid] = make_float4(mag * cosf(th), mag * sinf(th),
                            m128 * cosf(a128), m128 * sinf(a128));
  }
  for (int ent = tid; ent < 12 * 12 * 32; ent += nth) {
    int lane = ent & 31;
    int t2 = ent >> 5;
    int nt = t2 % 12, kt = t2 / 12;
    int n  = nt * 16 + (lane & 15);
    int kb = kt * 32 + (lane >> 4) * 8;
    __bf16* dst = Wp + ent * 16;
    float g = (n < 128) ? expf(gamma_log[n & 63]) : 1.f;
    for (int e = 0; e < 16; ++e) {
      int k = kb + (e & 7) + ((e >> 3) << 4);
      float v;
      if (n < 64)        v = B_re[n * DIN + k] * g;
      else if (n < 128)  v = B_im[(n - 64) * DIN + k] * g;
      else               v = Dm[(n - 128) * DIN + k];
      dst[e] = f2bf(v);
    }
  }
  for (int ent = tid; ent < 4 * 4 * 32; ent += nth) {
    int lane = ent & 31;
    int t2 = ent >> 5;
    int nt = t2 & 3, kt = t2 >> 2;
    int n  = nt * 16 + (lane & 15);
    int kb = kt * 32 + (lane >> 4) * 8;
    __bf16* dst = Cp + ent * 16;
    for (int e = 0; e < 16; ++e) {
      int k = kb + (e & 7) + ((e >> 3) << 4);
      float v = (k < 64) ? C_re[n * DH + k] : -C_im[n * DH + (k - 64)];
      dst[e] = f2bf(v);
    }
  }
}

__device__ __forceinline__ void loadB3(v16bf* Bf, const __bf16* __restrict__ Wp,
                                       int kt, int ntb, int lane) {
#pragma unroll
  for (int j = 0; j < 3; ++j)
    Bf[j] = *(const v16bf*)(Wp + (((kt * 12 + ntb + j) * 32 + lane) << 4));
}

// ---------------------------------------------------------------------------
// k_gemm: S(65536x192) = bf16(x) @ Wp.
// Block = 64 rows x 192 cols, 8 waves. Whole x-slab staged once in LDS
// (padded), then a barrier-free k-loop. Wave = 2 m-tiles x 3 n-tiles.
// BOTH A and B fragments ping-pong between two register sets, loaded one
// k-step ahead, so each 6-WMMA group runs behind prefetched operands.
// ---------------------------------------------------------------------------
__global__ void __launch_bounds__(256) k_gemm(const float* __restrict__ x,
                                              const __bf16* __restrict__ Wp,
                                              __bf16* __restrict__ S) {
  __shared__ __bf16 Xb[64 * XPAD];            // 50,176 B
  const int tid  = threadIdx.x;
  const int lane = tid & 31;
  const int wv   = tid >> 5;
  const long m0  = (long)blockIdx.x * 64;

  // ---- stage x slab (64 x 384 f32 -> bf16 LDS) ----
  {
    const int srow = tid >> 2;                // 0..63
    const int q    = tid & 3;                 // quarter of a row (96 floats)
    const float* xp = x + (size_t)(m0 + srow) * DIN + q * 96;
    __bf16* dst = &Xb[srow * XPAD + q * 96];
    __builtin_prefetch(xp, 0, 1);
#pragma unroll
    for (int i = 0; i < 12; ++i) {
      float4 f0 = ((const float4*)xp)[2 * i];
      float4 f1 = ((const float4*)xp)[2 * i + 1];
      union { __bf16 e[8]; v8bf v; } u;
      u.e[0] = f2bf(f0.x); u.e[1] = f2bf(f0.y); u.e[2] = f2bf(f0.z); u.e[3] = f2bf(f0.w);
      u.e[4] = f2bf(f1.x); u.e[5] = f2bf(f1.y); u.e[6] = f2bf(f1.z); u.e[7] = f2bf(f1.w);
      *(v8bf*)(dst + i * 8) = u.v;
    }
  }
  __syncthreads();

  // ---- wave tiling: mtp selects m-tile pair {2*mtp, 2*mtp+1}; ntb = 3-wide n group
  const int mtp  = wv & 1;
  const int ntb  = (wv >> 1) * 3;             // 0,3,6,9
  const int hsel = lane >> 4;
  const int arow0 = (mtp * 2) * 16 + (lane & 15);
  const int arow1 = arow0 + 16;
  const __bf16* a0base = &Xb[arow0 * XPAD];
  const __bf16* a1base = &Xb[arow1 * XPAD];

  const v8f vz = {0.f, 0.f, 0.f, 0.f, 0.f, 0.f, 0.f, 0.f};
  v8f acc[2][3];
#pragma unroll
  for (int m = 0; m < 2; ++m)
#pragma unroll
    for (int j = 0; j < 3; ++j) acc[m][j] = vz;

  v16bf Ba[3], Bb[3], A0a, A1a, A0b, A1b;
  loadB3(Ba, Wp, 0, ntb, lane);
  A0a = ldfragA(a0base, hsel);
  A1a = ldfragA(a1base, hsel);

  for (int ktp = 0; ktp < 6; ++ktp) {
    const int kt0 = 2 * ktp, kt1 = kt0 + 1;
    // prefetch odd-step operands into free buffers
    loadB3(Bb, Wp, kt1, ntb, lane);
    A0b = ldfragA(a0base + kt1 * 32, hsel);
    A1b = ldfragA(a1base + kt1 * 32, hsel);
#pragma unroll
    for (int j = 0; j < 3; ++j) {
      acc[0][j] = __builtin_amdgcn_wmma_f32_16x16x32_bf16(false, A0a, false, Ba[j],
                                                          (short)0, acc[0][j], false, false);
      acc[1][j] = __builtin_amdgcn_wmma_f32_16x16x32_bf16(false, A1a, false, Ba[j],
                                                          (short)0, acc[1][j], false, false);
    }
    if (ktp < 5) {                            // prefetch next even-step operands
      loadB3(Ba, Wp, kt0 + 2, ntb, lane);
      A0a = ldfragA(a0base + (kt0 + 2) * 32, hsel);
      A1a = ldfragA(a1base + (kt0 + 2) * 32, hsel);
    }
#pragma unroll
    for (int j = 0; j < 3; ++j) {
      acc[0][j] = __builtin_amdgcn_wmma_f32_16x16x32_bf16(false, A0b, false, Bb[j],
                                                          (short)0, acc[0][j], false, false);
      acc[1][j] = __builtin_amdgcn_wmma_f32_16x16x32_bf16(false, A1b, false, Bb[j],
                                                          (short)0, acc[1][j], false, false);
    }
  }

  // ---- epilogue: C layout lanes>=16 hold M=r+8 ----
  const int rbase = (lane >> 4) * 8;
#pragma unroll
  for (int m = 0; m < 2; ++m) {
    const long rown = m0 + (mtp * 2 + m) * 16 + rbase;
#pragma unroll
    for (int j = 0; j < 3; ++j) {
      const int col = (ntb + j) * 16 + (lane & 15);
#pragma unroll
      for (int rr = 0; rr < 8; ++rr)
        S[(size_t)(rown + rr) * SCOLS + col] = f2bf(acc[m][j][rr]);
    }
  }
}

// ---------------------------------------------------------------------------
// k_carry: per (batch,chunk) chunk-local scan carry. blockIdx.x = b*16+c.
// ---------------------------------------------------------------------------
__global__ void k_carry(const __bf16* __restrict__ S, const float4* __restrict__ lamT,
                        float2* __restrict__ car) {
  const int blk = blockIdx.x;
  const int h = threadIdx.x;
  float4 lt = lamT[h];
  const float lr = lt.x, li = lt.y;
  const size_t base = (size_t)blk * CHUNK * SCOLS;
  float sr = 0.f, si = 0.f;
#pragma unroll 4
  for (int t = 0; t < CHUNK; ++t) {
    float br = bf2f(S[base + (size_t)t * SCOLS + h]);
    float bi = bf2f(S[base + (size_t)t * SCOLS + 64 + h]);
    float nr = lr * sr - li * si + br;
    float ni = lr * si + li * sr + bi;
    sr = nr; si = ni;
  }
  car[blk * 64 + h] = make_float2(sr, si);
}

// ---------------------------------------------------------------------------
// k_scanproj: prefix from carries, local scan -> bf16 xs in LDS, then
// y = [xs_re|xs_im] @ Cp (WMMA K=128, B fragments preloaded in registers,
// A fragments prefetched one k-step ahead) + Du, column max over chunk.
// ---------------------------------------------------------------------------
__global__ void __launch_bounds__(256) k_scanproj(const __bf16* __restrict__ S,
                                                  const __bf16* __restrict__ Cp,
                                                  const float4* __restrict__ lamT,
                                                  const float2* __restrict__ car,
                                                  float* __restrict__ pmax) {
  __shared__ __bf16 XS[CHUNK * XSPAD];        // 34,816 B
  __shared__ float red[8 * 64];
  const int blk = blockIdx.x;
  const int b = blk >> 4, c = blk & 15;
  const int tid = threadIdx.x;
  const int lane = tid & 31;
  const int wv = tid >> 5;

  // Preload all 16 C fragments (4 kt x 4 nt); loads overlap the scan phase.
  v16bf CB[16];
#pragma unroll
  for (int q = 0; q < 16; ++q)
    CB[q] = *(const v16bf*)(Cp + ((q * 32 + lane) << 4));

  if (tid < 64) {                              // waves 0-1 fully active
    const int h = tid;
    float4 lt = lamT[h];
    const float lr = lt.x, li = lt.y, Ar = lt.z, Ai = lt.w;
    float sr = 0.f, si = 0.f;
    for (int j = 0; j < c; ++j) {              // prefix: combine prior chunk carries
      float2 cj = car[(b * 16 + j) * 64 + h];
      float nr = Ar * sr - Ai * si + cj.x;
      float ni = Ar * si + Ai * sr + cj.y;
      sr = nr; si = ni;
    }
    const size_t base = (size_t)blk * CHUNK * SCOLS;
#pragma unroll 4
    for (int t = 0; t < CHUNK; ++t) {
      float br = bf2f(S[base + (size_t)t * SCOLS + h]);
      float bi = bf2f(S[base + (size_t)t * SCOLS + 64 + h]);
      float nr = lr * sr - li * si + br;
      float ni = lr * si + li * sr + bi;
      sr = nr; si = ni;
      XS[t * XSPAD + h]      = f2bf(sr);
      XS[t * XSPAD + 64 + h] = f2bf(si);
    }
  }
  __syncthreads();

  const v8f vz = {0.f, 0.f, 0.f, 0.f, 0.f, 0.f, 0.f, 0.f};
  v8f acc[4] = {vz, vz, vz, vz};
  const int arow = wv * 16 + (lane & 15);
  const int hsel = lane >> 4;
  const __bf16* abase = &XS[arow * XSPAD];
  v16bf a = ldfragA(abase, hsel);
#pragma unroll
  for (int kt = 0; kt < 4; ++kt) {
    v16bf an;
    if (kt < 3) an = ldfragA(abase + (kt + 1) * 32, hsel);  // prefetch next A
#pragma unroll
    for (int nt = 0; nt < 4; ++nt)
      acc[nt] = __builtin_amdgcn_wmma_f32_16x16x32_bf16(false, a, false, CB[kt * 4 + nt],
                                                        (short)0, acc[nt], false, false);
    a = an;   // fully unrolled: SSA rename, no movs
  }

  const size_t rowbase = (size_t)blk * CHUNK;
  const int rbase = (lane >> 4) * 8;
#pragma unroll
  for (int nt = 0; nt < 4; ++nt) {
    const int o = nt * 16 + (lane & 15);
    float m = -3.402823466e38f;
#pragma unroll
    for (int rr = 0; rr < 8; ++rr) {
      size_t grow = rowbase + wv * 16 + rbase + rr;
      float du = bf2f(S[grow * SCOLS + 128 + o]);
      m = fmaxf(m, acc[nt][rr] + du);
    }
    m = fmaxf(m, __shfl_xor(m, 16, 32));       // merge rows held by lane l / l+16
    if (lane < 16) red[wv * 64 + o] = m;
  }
  __syncthreads();
  if (tid < 64) {
    float m = red[tid];
#pragma unroll
    for (int w2 = 1; w2 < 8; ++w2) m = fmaxf(m, red[w2 * 64 + tid]);
    pmax[blk * 64 + tid] = m;
  }
}

// ---------------------------------------------------------------------------
// k_reduce: max over 16 chunks -> (32, 64)
// ---------------------------------------------------------------------------
__global__ void k_reduce(const float* __restrict__ pmax, float* __restrict__ out) {
  int t = blockIdx.x * blockDim.x + threadIdx.x;  // 0..2047
  int b = t >> 6, o = t & 63;
  float m = -3.402823466e38f;
  for (int c = 0; c < NCHUNK; ++c) m = fmaxf(m, pmax[(b * 16 + c) * 64 + o]);
  out[b * 64 + o] = m;
}

extern "C" void kernel_launch(void* const* d_in, const int* in_sizes, int n_in,
                              void* d_out, int out_size, void* d_ws, size_t ws_size,
                              hipStream_t stream) {
  const float* x         = (const float*)d_in[0];
  const float* nu_log    = (const float*)d_in[1];
  const float* theta_log = (const float*)d_in[2];
  const float* gamma_log = (const float*)d_in[3];
  const float* B_re      = (const float*)d_in[4];
  const float* B_im      = (const float*)d_in[5];
  const float* C_re      = (const float*)d_in[6];
  const float* C_im      = (const float*)d_in[7];
  const float* Dm        = (const float*)d_in[8];

  char* ws = (char*)d_ws;
  __bf16* S    = (__bf16*)(ws + OFF_S);
  __bf16* Wp   = (__bf16*)(ws + OFF_W);
  __bf16* Cp   = (__bf16*)(ws + OFF_C);
  float4* lamT = (float4*)(ws + OFF_LAM);
  float2* car  = (float2*)(ws + OFF_CAR);
  float* pmax  = (float*)(ws + OFF_PMAX);
  float* out   = (float*)d_out;

  hipLaunchKernelGGL(k_prep, dim3(20), dim3(256), 0, stream,
                     nu_log, theta_log, gamma_log, B_re, B_im, C_re, C_im, Dm,
                     Wp, Cp, lamT);
  hipLaunchKernelGGL(k_gemm, dim3(NROW / 64), dim3(256), 0, stream, x, Wp, S);
  hipLaunchKernelGGL(k_carry, dim3(BATCHN * NCHUNK), dim3(64), 0, stream, S, lamT, car);
  hipLaunchKernelGGL(k_scanproj, dim3(BATCHN * NCHUNK), dim3(256), 0, stream,
                     S, Cp, lamT, car, pmax);
  hipLaunchKernelGGL(k_reduce, dim3(8), dim3(256), 0, stream, pmax, out);
}
